// DCCA_56023553409364
// MI455X (gfx1250) — matrix operand order repaired and checked by reading, not compiled, not run on hardware
//
#include <hip/hip_runtime.h>
#include <hip/hip_bf16.h>

// ---------------------------------------------------------------------------
// CDNA5 (gfx1250) implementation of dwconv -> 2x criss-cross attention -> 1x1
// All GEMM-shaped stages use v_wmma_f32_16x16x32_bf16 (wave32 WMMA).
// ---------------------------------------------------------------------------

typedef __bf16 bf16_t;
typedef __attribute__((ext_vector_type(16))) __bf16 v16bf;
typedef __attribute__((ext_vector_type(8)))  float  v8f;

#define HW_   16384   // 128*128
#define H_    128
#define W_    128
#define C_    64
#define B_    8

// ---- WMMA fragment loaders (contiguous k order, same for A and B) ------------------
// A-matrix 16x32 bf16 (MxK): lane m = lane&15; lanes 0-15 hold K 0..15,
// lanes 16-31 hold K 16..31.
__device__ __forceinline__ v16bf load_A_frag(const bf16_t* base, int ld, int kbase, int lane) {
    int m = lane & 15;
    int g = lane >> 4;
    const bf16_t* row = base + m * ld + kbase;
    v16bf a;
#pragma unroll
    for (int i = 0; i < 16; ++i) a[i] = row[g * 16 + i];
    return a;
}

// B-matrix 32x16 bf16 (KxN): lane n = lane&15; lanes 0-15 hold K 0..15,
// lanes 16-31 hold K 16..31 (element i -> K = (lane>>4)*16 + i).
__device__ __forceinline__ v16bf load_B_frag(const bf16_t* base, int ld, int kbase, int lane) {
    int n  = lane & 15;
    int kb = kbase + (lane >> 4) * 16;
    v16bf b;
#pragma unroll
    for (int i = 0; i < 16; ++i) b[i] = base[(kb + i) * ld + n];
    return b;
}

__device__ __forceinline__ v8f wmma_bf16(v16bf a, v16bf b, v8f c) {
    return __builtin_amdgcn_wmma_f32_16x16x32_bf16(false, a, false, b, (short)0, c, false, false);
}

// ---------------------------------------------------------------------------
// Depthwise 3x3 conv, zero padding.  grid = B*C blocks.
// ---------------------------------------------------------------------------
__global__ __launch_bounds__(256) void dwconv_kernel(const float* __restrict__ x,
                                                     const float* __restrict__ wdw,
                                                     float* __restrict__ y) {
    int bc = blockIdx.x;                  // 0..511
    int c  = bc & (C_ - 1);
    const float* xin  = x + (size_t)bc * HW_;
    float*       yout = y + (size_t)bc * HW_;
    float wgt[9];
#pragma unroll
    for (int i = 0; i < 9; ++i) wgt[i] = wdw[c * 9 + i];
    for (int p = threadIdx.x; p < HW_; p += 256) {
        int h = p >> 7, w = p & 127;
        float acc = 0.f;
#pragma unroll
        for (int dh = -1; dh <= 1; ++dh)
#pragma unroll
            for (int dw = -1; dw <= 1; ++dw) {
                int hh = h + dh, ww = w + dw;
                if (hh >= 0 && hh < H_ && ww >= 0 && ww < W_)
                    acc += xin[hh * W_ + ww] * wgt[(dh + 1) * 3 + (dw + 1)];
            }
        yout[p] = acc;
    }
}

// ---------------------------------------------------------------------------
// QKV projection. qk stacked as one 16x64 A-tile (rows 0-7 = wq, 8-15 = wk),
// v as 64x64 A. Each wave owns a 16-pixel N-tile. grid = BHW/128 blocks.
// ---------------------------------------------------------------------------
__global__ __launch_bounds__(256) void proj_qkv_kernel(const float* __restrict__ y,
                                                       const float* __restrict__ wq,
                                                       const float* __restrict__ wk,
                                                       const float* __restrict__ wv,
                                                       float* __restrict__ qb,
                                                       float* __restrict__ kb,
                                                       bf16_t* __restrict__ vb) {
    __shared__ bf16_t wqk[16][64];
    __shared__ bf16_t wvs[64][64];
    int t = threadIdx.x, lane = t & 31, wv_id = t >> 5;
    for (int i = t; i < 16 * 64; i += 256) {
        int r = i >> 6, c = i & 63;
        wqk[r][c] = (bf16_t)(r < 8 ? wq[r * 64 + c] : wk[(r - 8) * 64 + c]);
    }
    for (int i = t; i < 64 * 64; i += 256) wvs[i >> 6][i & 63] = (bf16_t)wv[i];
    __syncthreads();

    size_t p0  = (size_t)blockIdx.x * 128 + wv_id * 16;
    int    b   = (int)(p0 / HW_);
    int    hw0 = (int)(p0 % HW_);
    int    n   = lane & 15;
    int    koff = (lane >> 4) * 16;

    // B fragments: y channels for this wave's 16 pixels, two K=32 steps.
    v16bf Bf[2];
#pragma unroll
    for (int ks = 0; ks < 2; ++ks)
#pragma unroll
        for (int i = 0; i < 16; ++i) {
            int c = ks * 32 + koff + i;
            Bf[ks][i] = (bf16_t)y[((size_t)(b * C_ + c)) * HW_ + hw0 + n];
        }

    // q,k (one 16-row tile)
    {
        v8f acc = {};
#pragma unroll
        for (int ks = 0; ks < 2; ++ks)
            acc = wmma_bf16(load_A_frag(&wqk[0][0], 64, ks * 32, lane), Bf[ks], acc);
        int mo = (lane >> 4) * 8;
#pragma unroll
        for (int r = 0; r < 8; ++r) {
            int m = mo + r;
            if (m < 8) qb[((size_t)(b * 8 + m)) * HW_ + hw0 + n] = acc[r];
            else       kb[((size_t)(b * 8 + m - 8)) * HW_ + hw0 + n] = acc[r];
        }
    }
    // v (four 16-row tiles), stored bf16 for the attention GEMMs
#pragma unroll
    for (int mt = 0; mt < 4; ++mt) {
        v8f acc = {};
#pragma unroll
        for (int ks = 0; ks < 2; ++ks)
            acc = wmma_bf16(load_A_frag(&wvs[mt * 16][0], 64, ks * 32, lane), Bf[ks], acc);
        int mo = (lane >> 4) * 8;
#pragma unroll
        for (int r = 0; r < 8; ++r) {
            int d = mt * 16 + mo + r;
            vb[((size_t)(b * C_ + d)) * HW_ + hw0 + n] = (bf16_t)acc[r];
        }
    }
}

// ---------------------------------------------------------------------------
// Column attention pass (fixed b,w; attend over g=h, diag masked).
// Produces oH[b][w][h][64], mH, sH.  grid = B*W blocks.
// ---------------------------------------------------------------------------
__global__ __launch_bounds__(256) void cca_col_kernel(const float* __restrict__ q,
                                                      const float* __restrict__ k,
                                                      const bf16_t* __restrict__ v,
                                                      float* __restrict__ oH,
                                                      float* __restrict__ mHb,
                                                      float* __restrict__ sHb) {
    int blk = blockIdx.x, b = blk >> 7, w = blk & 127;
    int t = threadIdx.x, lane = t & 31, wvid = t >> 5;
    __shared__ float  qs[8][128];
    __shared__ float  ks_[8][128];
    __shared__ bf16_t vs[128][64];    // v^T: [g][d]
    __shared__ bf16_t Ps[128][128];   // logits then exp-weights (A operand)

    for (int i = t; i < 8 * 128; i += 256) {
        int d = i >> 7, h = i & 127;
        qs[d][h]  = q[((size_t)(b * 8 + d)) * HW_ + h * W_ + w];
        ks_[d][h] = k[((size_t)(b * 8 + d)) * HW_ + h * W_ + w];
    }
    for (int i = t; i < 128 * 64; i += 256) {
        int g = i >> 6, d = i & 63;
        vs[g][d] = v[((size_t)(b * C_ + d)) * HW_ + g * W_ + w];
    }
    __syncthreads();

    // logits eH[h][g] = sum_d q[d][h]*k[d][g] (K=8 -> VALU), diag masked
    {
        int h = t >> 1, g0 = (t & 1) * 64;
        float qv[8];
#pragma unroll
        for (int d = 0; d < 8; ++d) qv[d] = qs[d][h];
        for (int g = g0; g < g0 + 64; ++g) {
            float e = 0.f;
#pragma unroll
            for (int d = 0; d < 8; ++d) e += qv[d] * ks_[d][g];
            if (g == h) e = -1e30f;
            Ps[h][g] = (bf16_t)e;
        }
    }
    __syncthreads();

    // per-row softmax stats + exp weights
    if (t < 128) {
        int h = t;
        float m = -3.0e38f;
        for (int g = 0; g < 128; ++g) m = fmaxf(m, (float)Ps[h][g]);
        float s = 0.f;
        for (int g = 0; g < 128; ++g) {
            float p = __expf((float)Ps[h][g] - m);
            s += p;
            Ps[h][g] = (bf16_t)p;
        }
        mHb[(b * W_ + w) * H_ + h] = m;
        sHb[(b * W_ + w) * H_ + h] = s;
    }
    __syncthreads();

    // oH(128x64) = Ps(128x128) x vs(128x64), WMMA bf16
    int mt = wvid;                       // 8 waves -> 8 M-tiles
#pragma unroll
    for (int nt = 0; nt < 4; ++nt) {
        v8f acc = {};
#pragma unroll
        for (int ks4 = 0; ks4 < 4; ++ks4)
            acc = wmma_bf16(load_A_frag(&Ps[mt * 16][0], 128, ks4 * 32, lane),
                            load_B_frag(&vs[0][nt * 16], 64, ks4 * 32, lane), acc);
        int n = lane & 15, mo = (lane >> 4) * 8;
#pragma unroll
        for (int r = 0; r < 8; ++r) {
            int h = mt * 16 + mo + r;
            int d = nt * 16 + n;
            oH[(((size_t)(b * W_ + w)) * H_ + h) * 64 + d] = acc[r];
        }
    }
}

// ---------------------------------------------------------------------------
// Row attention pass (fixed b,h; attend over w'; no diag mask).
// Produces oW[b][h][w][64], mW, sW.  grid = B*H blocks.
// ---------------------------------------------------------------------------
__global__ __launch_bounds__(256) void cca_row_kernel(const float* __restrict__ q,
                                                      const float* __restrict__ k,
                                                      const bf16_t* __restrict__ v,
                                                      float* __restrict__ oW,
                                                      float* __restrict__ mWb,
                                                      float* __restrict__ sWb) {
    int blk = blockIdx.x, b = blk >> 7, h = blk & 127;
    int t = threadIdx.x, lane = t & 31, wvid = t >> 5;
    __shared__ float  qs[8][128];
    __shared__ float  ks_[8][128];
    __shared__ bf16_t vs[128][64];    // v^T: [w'][d]
    __shared__ bf16_t Ps[128][128];

    for (int i = t; i < 8 * 128; i += 256) {
        int d = i >> 7, w = i & 127;
        qs[d][w]  = q[((size_t)(b * 8 + d)) * HW_ + h * W_ + w];
        ks_[d][w] = k[((size_t)(b * 8 + d)) * HW_ + h * W_ + w];
    }
    for (int i = t; i < 128 * 64; i += 256) {
        int wp = i >> 6, d = i & 63;
        vs[wp][d] = v[((size_t)(b * C_ + d)) * HW_ + h * W_ + wp];
    }
    __syncthreads();

    {
        int w = t >> 1, g0 = (t & 1) * 64;
        float qv[8];
#pragma unroll
        for (int d = 0; d < 8; ++d) qv[d] = qs[d][w];
        for (int g = g0; g < g0 + 64; ++g) {
            float e = 0.f;
#pragma unroll
            for (int d = 0; d < 8; ++d) e += qv[d] * ks_[d][g];
            Ps[w][g] = (bf16_t)e;
        }
    }
    __syncthreads();

    if (t < 128) {
        int w = t;
        float m = -3.0e38f;
        for (int g = 0; g < 128; ++g) m = fmaxf(m, (float)Ps[w][g]);
        float s = 0.f;
        for (int g = 0; g < 128; ++g) {
            float p = __expf((float)Ps[w][g] - m);
            s += p;
            Ps[w][g] = (bf16_t)p;
        }
        mWb[(b * H_ + h) * W_ + w] = m;
        sWb[(b * H_ + h) * W_ + w] = s;
    }
    __syncthreads();

    int mt = wvid;
#pragma unroll
    for (int nt = 0; nt < 4; ++nt) {
        v8f acc = {};
#pragma unroll
        for (int ks4 = 0; ks4 < 4; ++ks4)
            acc = wmma_bf16(load_A_frag(&Ps[mt * 16][0], 128, ks4 * 32, lane),
                            load_B_frag(&vs[0][nt * 16], 64, ks4 * 32, lane), acc);
        int n = lane & 15, mo = (lane >> 4) * 8;
#pragma unroll
        for (int r = 0; r < 8; ++r) {
            int w = mt * 16 + mo + r;
            int d = nt * 16 + n;
            oW[(((size_t)(b * H_ + h)) * W_ + w) * 64 + d] = acc[r];
        }
    }
}

// ---------------------------------------------------------------------------
// Exact softmax merge of the two passes + gamma * (.) + residual.
// grid = BHW/256 blocks.
// ---------------------------------------------------------------------------
__global__ __launch_bounds__(256) void cca_combine_kernel(const float* __restrict__ yin,
                                                          const float* __restrict__ oH,
                                                          const float* __restrict__ oW,
                                                          const float* __restrict__ mHb,
                                                          const float* __restrict__ sHb,
                                                          const float* __restrict__ mWb,
                                                          const float* __restrict__ sWb,
                                                          const float* __restrict__ gamma,
                                                          float* __restrict__ yout) {
    int p = blockIdx.x * 256 + threadIdx.x;     // 0..BHW-1
    int b = p / HW_, hw = p % HW_;
    int h = hw >> 7, w = hw & 127;
    float mH = mHb[(b * W_ + w) * H_ + h];
    float sH = sHb[(b * W_ + w) * H_ + h];
    float mW = mWb[(b * H_ + h) * W_ + w];
    float sW = sWb[(b * H_ + h) * W_ + w];
    float m  = fmaxf(mH, mW);
    float aH = __expf(mH - m), aW = __expf(mW - m);
    float inv = 1.0f / (sH * aH + sW * aW);
    float g = gamma[0];
    const float* pH = &oH[(((size_t)(b * W_ + w)) * H_ + h) * 64];
    const float* pW = &oW[(((size_t)(b * H_ + h)) * W_ + w) * 64];
#pragma unroll 4
    for (int d = 0; d < C_; ++d) {
        float o = (pH[d] * aH + pW[d] * aW) * inv;
        size_t yi = ((size_t)(b * C_ + d)) * HW_ + hw;
        yout[yi] = g * o + yin[yi];
    }
}

// ---------------------------------------------------------------------------
// Final pointwise (1x1) conv: out = w_pw (64x64) x y.  grid = BHW/128 blocks.
// ---------------------------------------------------------------------------
__global__ __launch_bounds__(256) void pointwise_kernel(const float* __restrict__ y,
                                                        const float* __restrict__ wpw,
                                                        float* __restrict__ out) {
    __shared__ bf16_t ws[64][64];
    int t = threadIdx.x, lane = t & 31, wvid = t >> 5;
    for (int i = t; i < 64 * 64; i += 256) ws[i >> 6][i & 63] = (bf16_t)wpw[i];
    __syncthreads();

    size_t p0  = (size_t)blockIdx.x * 128 + wvid * 16;
    int    b   = (int)(p0 / HW_);
    int    hw0 = (int)(p0 % HW_);
    int    n   = lane & 15;
    int    koff = (lane >> 4) * 16;

    v16bf Bf[2];
#pragma unroll
    for (int ks = 0; ks < 2; ++ks)
#pragma unroll
        for (int i = 0; i < 16; ++i) {
            int c = ks * 32 + koff + i;
            Bf[ks][i] = (bf16_t)y[((size_t)(b * C_ + c)) * HW_ + hw0 + n];
        }

#pragma unroll
    for (int mt = 0; mt < 4; ++mt) {
        v8f acc = {};
#pragma unroll
        for (int ks = 0; ks < 2; ++ks)
            acc = wmma_bf16(load_A_frag(&ws[mt * 16][0], 64, ks * 32, lane), Bf[ks], acc);
        int mo = (lane >> 4) * 8;
#pragma unroll
        for (int r = 0; r < 8; ++r) {
            int o = mt * 16 + mo + r;
            out[((size_t)(b * C_ + o)) * HW_ + hw0 + n] = acc[r];
        }
    }
}

// ---------------------------------------------------------------------------
// Host launcher
// ---------------------------------------------------------------------------
extern "C" void kernel_launch(void* const* d_in, const int* in_sizes, int n_in,
                              void* d_out, int out_size, void* d_ws, size_t ws_size,
                              hipStream_t stream) {
    const float* x     = (const float*)d_in[0];
    const float* w_dw  = (const float*)d_in[1];
    const float* wq    = (const float*)d_in[2];
    const float* wk    = (const float*)d_in[3];
    const float* wv    = (const float*)d_in[4];
    const float* gamma = (const float*)d_in[5];
    const float* w_pw  = (const float*)d_in[6];
    float* out = (float*)d_out;

    const size_t Ny   = (size_t)B_ * C_ * HW_;       // 8.39M floats
    const size_t Nq   = (size_t)B_ * 8 * HW_;        // 1.05M floats
    const size_t Npx  = (size_t)B_ * HW_;            // 131072

    char* ws = (char*)d_ws;
    size_t off = 0;
    float*  y0   = (float*)(ws + off);  off += Ny * 4;
    float*  y1   = (float*)(ws + off);  off += Ny * 4;
    float*  qb   = (float*)(ws + off);  off += Nq * 4;
    float*  kb   = (float*)(ws + off);  off += Nq * 4;
    bf16_t* vb   = (bf16_t*)(ws + off); off += Ny * 2;
    float*  oHb  = (float*)(ws + off);  off += Ny * 4;   // [B][W][H][64]
    float*  oWb  = (float*)(ws + off);  off += Ny * 4;   // [B][H][W][64]
    float*  mHb  = (float*)(ws + off);  off += Npx * 4;
    float*  sHb  = (float*)(ws + off);  off += Npx * 4;
    float*  mWb  = (float*)(ws + off);  off += Npx * 4;
    float*  sWb  = (float*)(ws + off);  off += Npx * 4;
    (void)ws_size;

    const dim3 blk(256);

    // 1) depthwise conv
    dwconv_kernel<<<dim3(B_ * C_), blk, 0, stream>>>(x, w_dw, y0);

    // 2) CCA pass 1: y0 -> y1
    proj_qkv_kernel<<<dim3((B_ * HW_) / 128), blk, 0, stream>>>(y0, wq, wk, wv, qb, kb, vb);
    cca_col_kernel<<<dim3(B_ * W_), blk, 0, stream>>>(qb, kb, vb, oHb, mHb, sHb);
    cca_row_kernel<<<dim3(B_ * H_), blk, 0, stream>>>(qb, kb, vb, oWb, mWb, sWb);
    cca_combine_kernel<<<dim3((B_ * HW_) / 256), blk, 0, stream>>>(y0, oHb, oWb, mHb, sHb, mWb, sWb, gamma, y1);

    // 3) CCA pass 2: y1 -> y0
    proj_qkv_kernel<<<dim3((B_ * HW_) / 128), blk, 0, stream>>>(y1, wq, wk, wv, qb, kb, vb);
    cca_col_kernel<<<dim3(B_ * W_), blk, 0, stream>>>(qb, kb, vb, oHb, mHb, sHb);
    cca_row_kernel<<<dim3(B_ * H_), blk, 0, stream>>>(qb, kb, vb, oWb, mWb, sWb);
    cca_combine_kernel<<<dim3((B_ * HW_) / 256), blk, 0, stream>>>(y1, oHb, oWb, mHb, sHb, mWb, sWb, gamma, y0);

    // 4) pointwise conv -> output
    pointwise_kernel<<<dim3((B_ * HW_) / 128), blk, 0, stream>>>(y0, w_pw, out);
}
